// WindowAttention3D_9964324126868
// MI455X (gfx1250) — matrix operand order repaired and checked by reading, not compiled, not run on hardware
//
#include <hip/hip_runtime.h>
#include <hip/hip_fp16.h>

typedef __attribute__((ext_vector_type(16))) _Float16 v16h;
typedef __attribute__((ext_vector_type(8)))  float    v8f;
typedef __attribute__((ext_vector_type(4)))  float    v4f;

#define NH     6
#define HD     32
#define NTOK   256      // tokens per window
#define CCH    192
#define LTOT   65536
#define LN_EPS 1e-5f

// ---- LDS layout (bytes). Total 310,944 <= 320KB/workgroup on CDNA5 WGP. ----
#define SK_STR   40                       // halves per K row (32 + 8 pad -> conflict-free b128 B-frags)
#define SVT_STR  264                      // halves per V^T row (256 + 8 pad)
#define SS_STR   260                      // floats per score row (256 + 4 pad; keeps 16B align + bank spread)
#define SK_OFF   0                        // 256*40*2      = 20480
#define SVT_OFF  20480                    // 32*264*2      = 16896
#define SPOS_OFF 37376                    // 1576*4        =  6304
#define SS_OFF   43680                    // 256*260*4     = 266240
#define SINV_OFF 309920                   // 256*4         =  1024
#define LDS_BYTES 310944

// ---------------- Kernel 1: CPB MLP -> pos table (1575 x 6) ----------------
__device__ __forceinline__ void ln_relu12(float* h, const float* g, const float* b) {
  float mu = 0.f;
  #pragma unroll
  for (int j = 0; j < 12; ++j) mu += h[j];
  mu *= (1.f / 12.f);
  float var = 0.f;
  #pragma unroll
  for (int j = 0; j < 12; ++j) { float d = h[j] - mu; var += d * d; }
  var *= (1.f / 12.f);
  const float rs = rsqrtf(var + LN_EPS);
  #pragma unroll
  for (int j = 0; j < 12; ++j) {
    float t = (h[j] - mu) * rs * g[j] + b[j];
    h[j] = t > 0.f ? t : 0.f;
  }
}

__global__ void cpb_mlp_kernel(const float* __restrict__ w0, const float* __restrict__ b0,
                               const float* __restrict__ g1, const float* __restrict__ be1,
                               const float* __restrict__ w1, const float* __restrict__ b1,
                               const float* __restrict__ g2, const float* __restrict__ be2,
                               const float* __restrict__ w2, const float* __restrict__ b2,
                               const float* __restrict__ g3, const float* __restrict__ be3,
                               const float* __restrict__ w3, const float* __restrict__ b3,
                               float* __restrict__ posOut) {
  const int i = blockIdx.x * blockDim.x + threadIdx.x;
  if (i >= 1575) return;
  const float x0 = (float)(i / 225 - 3);        // bd in [-3,3]
  const float x1 = (float)((i / 15) % 15 - 7);  // bh in [-7,7]
  const float x2 = (float)(i % 15 - 7);         // bw in [-7,7]
  float h[12], y[12];
  #pragma unroll
  for (int j = 0; j < 12; ++j)
    h[j] = x0 * w0[j * 3 + 0] + x1 * w0[j * 3 + 1] + x2 * w0[j * 3 + 2] + b0[j];
  ln_relu12(h, g1, be1);
  #pragma unroll
  for (int j = 0; j < 12; ++j) {
    float s = b1[j];
    #pragma unroll
    for (int k = 0; k < 12; ++k) s += h[k] * w1[j * 12 + k];
    y[j] = s;
  }
  ln_relu12(y, g2, be2);
  #pragma unroll
  for (int j = 0; j < 12; ++j) {
    float s = b2[j];
    #pragma unroll
    for (int k = 0; k < 12; ++k) s += y[k] * w2[j * 12 + k];
    h[j] = s;
  }
  ln_relu12(h, g3, be3);
  #pragma unroll
  for (int o = 0; o < NH; ++o) {
    float s = b3[o];
    #pragma unroll
    for (int k = 0; k < 12; ++k) s += h[k] * w3[o * 12 + k];
    posOut[i * NH + o] = s;
  }
}

// -------- Kernel 2: one block per (window, head); WMMA f16 attention --------
extern __shared__ char smem[];

__global__ __launch_bounds__(256)
void win_attn_kernel(const float* __restrict__ qkv, const float* __restrict__ pos,
                     float* __restrict__ out) {
  const int head = blockIdx.x % NH;
  const int win  = blockIdx.x / NH;
  const int wd = win >> 6, wh = (win >> 3) & 7, ww = win & 7;
  const int tid  = threadIdx.x;
  const int lane = tid & 31;
  const int wave = tid >> 5;

  _Float16* sK  = (_Float16*)(smem + SK_OFF);    // [256][SK_STR] f16 (row-major K)
  _Float16* sVt = (_Float16*)(smem + SVT_OFF);   // [32][SVT_STR] f16 (V transposed)
  float*    sPos= (float*)  (smem + SPOS_OFF);   // [1575] bias for this head
  float*    sS  = (float*)  (smem + SS_OFF);     // [256][SS_STR] f32 scores / f16 exp
  float*    sInv= (float*)  (smem + SINV_OFF);   // [256] 1/rowsum

  auto gtok = [wd, wh, ww](int t) -> int {       // window token -> global token index
    const int d = t >> 6, h = (t >> 3) & 7, w = t & 7;
    return ((wd * 4 + d) * 64 + (wh * 8 + h)) * 64 + (ww * 8 + w);
  };

  // ---- Stage K (row-major) and V (transposed) as f16; bias column to LDS ----
  {
    const int t = tid;
    const int g = gtok(t);
    const float* kp = qkv + (size_t)LTOT * CCH     + (size_t)g * CCH + head * HD;
    const float* vp = qkv + (size_t)2 * LTOT * CCH + (size_t)g * CCH + head * HD;
    _Float16* kr = sK + t * SK_STR;
    #pragma unroll
    for (int c4 = 0; c4 < 8; ++c4) {
      const v4f k4 = *(const v4f*)(kp + c4 * 4);
      const v4f vv = *(const v4f*)(vp + c4 * 4);
      #pragma unroll
      for (int j = 0; j < 4; ++j) {
        kr[c4 * 4 + j] = (_Float16)k4[j];
        sVt[(c4 * 4 + j) * SVT_STR + t] = (_Float16)vv[j];
      }
    }
    for (int i = tid; i < 1575; i += 256) sPos[i] = pos[i * NH + head];
  }
  __syncthreads();

  // ---- Phase 1: S = (Q*scale) @ K^T + rpb  (bias rides the WMMA C operand) ----
  const float scale = 0.17677669529663687f; // 32^-0.5
  for (int hlf = 0; hlf < 2; ++hlf) {
    const int rt = wave + hlf * 8;          // row tile 0..15 (16 rows each)
    // A fragment (Q rows) loaded straight from global in WMMA A-layout:
    // lane<16 supplies K dims {0..7,16..23}, lane>=16 {8..15,24..31} of row lane&15.
    const int qrow = rt * 16 + (lane & 15);
    const float* qp = qkv + (size_t)gtok(qrow) * CCH + head * HD + (lane < 16 ? 0 : 8);
    const v4f q0 = *(const v4f*)(qp + 0);
    const v4f q1 = *(const v4f*)(qp + 4);
    const v4f q2 = *(const v4f*)(qp + 16);
    const v4f q3 = *(const v4f*)(qp + 20);
    union { v16h h; _Float16 e[16]; } ua;
    #pragma unroll
    for (int j = 0; j < 4; ++j) {
      ua.e[j]      = (_Float16)(q0[j] * scale);
      ua.e[4 + j]  = (_Float16)(q1[j] * scale);
      ua.e[8 + j]  = (_Float16)(q2[j] * scale);
      ua.e[12 + j] = (_Float16)(q3[j] * scale);
    }
    const v16h afrag = ua.h;
    const int nb = rt * 16 + (lane < 16 ? 0 : 8);
    // Gather index factorization: idx = lin(n) - lin(m) + 787  (787 = 3*225+7*15+7)
    int linN[8];
    #pragma unroll
    for (int r = 0; r < 8; ++r) {
      const int n = nb + r;
      linN[r] = (n >> 6) * 225 + ((n >> 3) & 7) * 15 + (n & 7) + 787;
    }
    #pragma unroll
    for (int ct = 0; ct < 16; ++ct) {
      const int m    = ct * 16 + (lane & 15);
      const int linM = (m >> 6) * 225 + ((m >> 3) & 7) * 15 + (m & 7);
      // Bias tile in C layout: VGPR r -> (row nb+r, col m). Loads batch, one wait.
      v8f cin;
      #pragma unroll
      for (int r = 0; r < 8; ++r) cin[r] = sPos[linN[r] - linM];
      // B fragment: column token = ct*16 + (lane&15); K dims {0..7,16..23} (lane<16) / {8..15,24..31}.
      const char* bp = (const char*)(sK + (ct * 16 + (lane & 15)) * SK_STR) + (lane < 16 ? 0 : 16);
      union { v16h h; v4f f[2]; } ub;
      ub.f[0] = *(const v4f*)bp;
      ub.f[1] = *(const v4f*)(bp + 32);
      const v8f d = __builtin_amdgcn_wmma_f32_16x16x32_f16(false, afrag, false, ub.h,
                                                           (short)0, cin, false, false);
      #pragma unroll
      for (int r = 0; r < 8; ++r) sS[(nb + r) * SS_STR + m] = d[r];
    }
  }
  __syncthreads();

  // ---- Phase 2: row softmax (1 thread/row). Store UNNORMALIZED exp as f16, ----
  // ---- front-packed in place; 1/sum goes to sInv and is applied in PV.      ----
  {
    float* rowp = sS + tid * SS_STR;
    const v4f* r4 = (const v4f*)rowp;
    v4f* w4 = (v4f*)rowp;
    float mx = -3.0e38f;
    #pragma unroll 8
    for (int j = 0; j < 64; ++j) {
      const v4f x = r4[j];
      mx = fmaxf(mx, fmaxf(fmaxf(x[0], x[1]), fmaxf(x[2], x[3])));
    }
    float sum = 0.f;
    for (int cch = 0; cch < 8; ++cch) {           // 32 floats per chunk
      v4f x[8];
      #pragma unroll
      for (int t = 0; t < 8; ++t) x[t] = r4[cch * 8 + t];
      asm volatile("" ::: "memory");              // keep packed writes after these reads
      union { _Float16 h[32]; v4f f[4]; } u;
      #pragma unroll
      for (int t = 0; t < 8; ++t) {
        #pragma unroll
        for (int j = 0; j < 4; ++j) {
          const float e = __expf(x[t][j] - mx);
          sum += e;
          u.h[t * 4 + j] = (_Float16)e;
        }
      }
      #pragma unroll
      for (int i = 0; i < 4; ++i) w4[cch * 4 + i] = u.f[i];  // bytes [64*cch, 64*cch+64)
    }
    sInv[tid] = 1.f / sum;
  }
  __syncthreads();

  // ---- Phase 3: out = (exp @ V) * (1/rowsum)  (exp rows already in A-layout) ----
  for (int hlf = 0; hlf < 2; ++hlf) {
    const int rt = wave + hlf * 8;
    v8f acc0 = {}, acc1 = {};
    const char* arow = (const char*)(sS + (rt * 16 + (lane & 15)) * SS_STR);
    #pragma unroll
    for (int kc = 0; kc < 8; ++kc) {              // K chunks of 32 attended tokens
      const char* ap = arow + kc * 64 + (lane < 16 ? 0 : 32);
      union { v16h h; v4f f[2]; } uA;
      uA.f[0] = *(const v4f*)ap;
      uA.f[1] = *(const v4f*)(ap + 16);
      const char* bp0 = (const char*)(sVt + ((lane & 15))      * SVT_STR) + kc * 64 + (lane < 16 ? 0 : 32);
      const char* bp1 = (const char*)(sVt + (16 + (lane & 15)) * SVT_STR) + kc * 64 + (lane < 16 ? 0 : 32);
      union { v16h h; v4f f[2]; } uB0, uB1;
      uB0.f[0] = *(const v4f*)bp0; uB0.f[1] = *(const v4f*)(bp0 + 16);
      uB1.f[0] = *(const v4f*)bp1; uB1.f[1] = *(const v4f*)(bp1 + 16);
      acc0 = __builtin_amdgcn_wmma_f32_16x16x32_f16(false, uA.h, false, uB0.h,
                                                    (short)0, acc0, false, false);
      acc1 = __builtin_amdgcn_wmma_f32_16x16x32_f16(false, uA.h, false, uB1.h,
                                                    (short)0, acc1, false, false);
    }
    const int nb = rt * 16 + (lane < 16 ? 0 : 8);
    float inv[8];
    #pragma unroll
    for (int r = 0; r < 8; ++r) inv[r] = sInv[nb + r];   // broadcast reads, conflict-free
    #pragma unroll
    for (int r = 0; r < 8; ++r) {
      float* op = out + (size_t)gtok(nb + r) * CCH + head * HD + (lane & 15);
      op[0]  = acc0[r] * inv[r];   // head dims 0..15
      op[16] = acc1[r] * inv[r];   // head dims 16..31
    }
  }
}

extern "C" void kernel_launch(void* const* d_in, const int* in_sizes, int n_in,
                              void* d_out, int out_size, void* d_ws, size_t ws_size,
                              hipStream_t stream) {
  (void)in_sizes; (void)n_in; (void)out_size; (void)ws_size;
  const float* qkv = (const float*)d_in[0];
  const float* w0  = (const float*)d_in[1];
  const float* b0  = (const float*)d_in[2];
  const float* g1  = (const float*)d_in[3];
  const float* be1 = (const float*)d_in[4];
  const float* w1  = (const float*)d_in[5];
  const float* b1  = (const float*)d_in[6];
  const float* g2  = (const float*)d_in[7];
  const float* be2 = (const float*)d_in[8];
  const float* w2  = (const float*)d_in[9];
  const float* b2  = (const float*)d_in[10];
  const float* g3  = (const float*)d_in[11];
  const float* be3 = (const float*)d_in[12];
  const float* w3  = (const float*)d_in[13];
  const float* b3  = (const float*)d_in[14];
  float* posWs = (float*)d_ws;        // 1575*6 floats
  float* outP  = (float*)d_out;

  // CDNA5 allows up to 320KB LDS per workgroup; raise the dynamic-LDS cap.
  (void)hipFuncSetAttribute(reinterpret_cast<const void*>(win_attn_kernel),
                            hipFuncAttributeMaxDynamicSharedMemorySize, LDS_BYTES);

  cpb_mlp_kernel<<<(1575 + 127) / 128, 128, 0, stream>>>(
      w0, b0, g1, be1, w1, b1, g2, be2, w2, b2, g3, be3, w3, b3, posWs);
  win_attn_kernel<<<256 * NH, 256, LDS_BYTES, stream>>>(qkv, posWs, outP);
}
